// ScaledDotProductAttention_36490042147144
// MI455X (gfx1250) — matrix operand order repaired and checked by reading, not compiled, not run on hardware
//
#include <hip/hip_runtime.h>

typedef _Float16 h16;
typedef __attribute__((ext_vector_type(4)))  float    v4f;
typedef __attribute__((ext_vector_type(8)))  float    v8f;
typedef __attribute__((ext_vector_type(8)))  _Float16 v8h;
typedef __attribute__((ext_vector_type(16))) _Float16 v16h;
typedef __attribute__((ext_vector_type(4)))  int      v4i;

#define S_LEN 2048
#define DKDIM 64
#define NBH   16      // B*H
#define BM    64      // query rows per workgroup
#define BN    64      // keys per tile
#define NT    (S_LEN / BN)
#define NWAVE 4
#define LDK   72      // padded LDS stride in halves (144B, 16B-aligned, bank-spread)

// ---- async global->LDS DMA (gfx1250) ----
#if defined(__HIP_DEVICE_COMPILE__) && __has_builtin(__builtin_amdgcn_global_load_async_to_lds_b128)
#define USE_ASYNC_LDS 1
#else
#define USE_ASYNC_LDS 0
#endif

#if USE_ASYNC_LDS
// param 0 is 'int4 __device__*' (probe-confirmed by clang diagnostic); LDS side AS3.
typedef v4i __attribute__((address_space(1)))* gb128_p;
typedef v4i __attribute__((address_space(3)))* lb128_p;
#if __has_builtin(__builtin_amdgcn_s_wait_asynccnt)
#define WAIT_ASYNC(n) __builtin_amdgcn_s_wait_asynccnt(n)
#else
#define WAIT_ASYNC(n) asm volatile("s_wait_asynccnt %0" :: "i"(n) : "memory")
#endif
#define SCRATCH_BYTES 32768
#else
#define WAIT_ASYNC(n) ((void)0)
#define SCRATCH_BYTES (NWAVE * 16 * LDK * 2)
#endif

// 16-lane butterfly reductions via DPP16 (each 16-lane half reduces independently):
// xor1 = quad_perm(1,0,3,2)=0xB1, xor2 = quad_perm(2,3,0,1)=0x4E,
// then row_half_mirror(0x141) (lane^7) and row_mirror(0x140) (lane^15).
__device__ __forceinline__ float red16_max(float v) {
  v = fmaxf(v, __int_as_float(__builtin_amdgcn_update_dpp(0, __float_as_int(v), 0xB1, 0xF, 0xF, true)));
  v = fmaxf(v, __int_as_float(__builtin_amdgcn_update_dpp(0, __float_as_int(v), 0x4E, 0xF, 0xF, true)));
  v = fmaxf(v, __int_as_float(__builtin_amdgcn_update_dpp(0, __float_as_int(v), 0x141, 0xF, 0xF, true)));
  v = fmaxf(v, __int_as_float(__builtin_amdgcn_update_dpp(0, __float_as_int(v), 0x140, 0xF, 0xF, true)));
  return v;
}
__device__ __forceinline__ float red16_sum(float v) {
  v += __int_as_float(__builtin_amdgcn_update_dpp(0, __float_as_int(v), 0xB1, 0xF, 0xF, true));
  v += __int_as_float(__builtin_amdgcn_update_dpp(0, __float_as_int(v), 0x4E, 0xF, 0xF, true));
  v += __int_as_float(__builtin_amdgcn_update_dpp(0, __float_as_int(v), 0x141, 0xF, 0xF, true));
  v += __int_as_float(__builtin_amdgcn_update_dpp(0, __float_as_int(v), 0x140, 0xF, 0xF, true));
  return v;
}

// Copy one 64-row x 128B f16 tile (global row stride grow_bytes) into padded LDS (144B rows).
__device__ __forceinline__ void dma_tile(const h16* gbase, size_t grow_bytes,
                                         _Float16* lds, int t) {
#pragma unroll
  for (int i = 0; i < 4; ++i) {
    const int idx = i * 128 + t;              // 512 16B segments / 128 threads
    const int row = idx >> 3, seg = (idx & 7) * 16;
#if USE_ASYNC_LDS
    __builtin_amdgcn_global_load_async_to_lds_b128(
        (gb128_p)((uintptr_t)gbase + (size_t)row * grow_bytes + seg),
        (lb128_p)(unsigned)(uintptr_t)((char*)lds + row * (LDK * 2) + seg), 0, 0);
#else
    v8h h = *(const v8h*)((const char*)gbase + (size_t)row * grow_bytes + seg);
    *(v8h*)((char*)lds + row * (LDK * 2) + seg) = h;
#endif
  }
}

// ============ pre-pass: K -> f16 [bh][key][d]; V -> f16 transposed [bh][d][key] ============
__global__ __launch_bounds__(128) void convert_kv(
    const float* __restrict__ K, const float* __restrict__ V,
    h16* __restrict__ Kh, h16* __restrict__ Vth)
{
  __shared__ __align__(32) _Float16 tr[DKDIM * LDK];
  const int t   = threadIdx.x;
  const int kt  = blockIdx.x & (NT - 1);
  const int bh  = blockIdx.x >> 5;
  const int row = t >> 1, d0 = (t & 1) * 32;
  const float* Kb = K + ((size_t)bh * S_LEN + kt * BN) * DKDIM;
  const float* Vb = V + ((size_t)bh * S_LEN + kt * BN) * DKDIM;
  // K: coalesced convert
  {
    const v4f* g = (const v4f*)(Kb + row * DKDIM + d0);
    h16* dst = Kh + (size_t)bh * S_LEN * DKDIM + (size_t)(kt * BN + row) * DKDIM + d0;
#pragma unroll
    for (int i = 0; i < 4; ++i) {
      v4f a = g[2 * i], b = g[2 * i + 1];
      v8h h;
#pragma unroll
      for (int e = 0; e < 4; ++e) { h[e] = (h16)a[e]; h[4 + e] = (h16)b[e]; }
      ((v8h*)dst)[i] = h;
    }
  }
  // V: coalesced row loads -> padded LDS transpose -> coalesced Vth row writes
  {
    const v4f* g = (const v4f*)(Vb + row * DKDIM + d0);
#pragma unroll
    for (int i = 0; i < 8; ++i) {
      v4f a = g[i];
#pragma unroll
      for (int e = 0; e < 4; ++e) tr[(d0 + 4 * i + e) * LDK + row] = (h16)a[e];
    }
  }
  __syncthreads();
  {
    const _Float16* src = &tr[row * LDK + d0];  // row = d, cols = key
    h16* dst = Vth + (size_t)bh * DKDIM * S_LEN + (size_t)row * S_LEN + kt * BN + d0;
#pragma unroll
    for (int i = 0; i < 4; ++i) ((v8h*)dst)[i] = *(const v8h*)(src + 8 * i);
  }
}

// ============ main: double-buffered async-DMA flash attention (f16 tiles) ============
__global__ __launch_bounds__(128) void trapz_flash_attn_pipe(
    const float* __restrict__ Q, const h16* __restrict__ Kh,
    const h16* __restrict__ Vth, const float* __restrict__ C,
    float* __restrict__ O)
{
  __shared__ __align__(32) _Float16 sK[2][BN * LDK];     // K tile [key][d]
  __shared__ __align__(32) _Float16 sV[2][DKDIM * LDK];  // V tile [d][key]
  __shared__ __align__(32) _Float16 sP[NWAVE][16 * LDK]; // per-wave P tiles
  __shared__ float sW[S_LEN];

  const int t    = threadIdx.x;
  const int wave = t >> 5;
  const int lane = t & 31;
  const int lhi  = lane >> 4;
  const int llo  = lane & 15;

  const int qt = blockIdx.x & (S_LEN / BM - 1);
  const int bh = blockIdx.x / (S_LEN / BM);

  const float* Qb  = Q   + (size_t)bh * S_LEN * DKDIM;
  const h16*   Kbh = Kh  + (size_t)bh * S_LEN * DKDIM;
  const h16*   Vbh = Vth + (size_t)bh * DKDIM * S_LEN;
  float*       Ob  = O   + (size_t)bh * S_LEN * DKDIM;

  for (int s = t; s < S_LEN; s += 128) {
    float cm = (s > 0)         ? fabsf(C[s] - C[s - 1]) : 0.f;
    float cp = (s < S_LEN - 1) ? fabsf(C[s + 1] - C[s]) : 0.f;
    sW[s] = 0.5f * (cm + cp);
  }

  // Q A-fragments (16x32 f16): lane row M=llo; elem j -> d = 32c + ((j&8)<<1) + 8*lhi + (j&7)
  const int qrow = qt * BM + wave * 16 + llo;
  v16h aq[2];
#pragma unroll
  for (int c = 0; c < 2; ++c) {
    const float* qp = Qb + qrow * DKDIM + 32 * c + 8 * lhi;
    v4f f0 = *(const v4f*)(qp);
    v4f f1 = *(const v4f*)(qp + 4);
    v4f f2 = *(const v4f*)(qp + 16);
    v4f f3 = *(const v4f*)(qp + 20);
#pragma unroll
    for (int i = 0; i < 4; ++i) {
      aq[c][i]      = (h16)f0[i];
      aq[c][4 + i]  = (h16)f1[i];
      aq[c][8 + i]  = (h16)f2[i];
      aq[c][12 + i] = (h16)f3[i];
    }
  }

  v8f oacc[4];
  float m[8], l[8];
#pragma unroll
  for (int i = 0; i < 4; ++i) oacc[i] = (v8f)(0.f);
#pragma unroll
  for (int r = 0; r < 8; ++r) { m[r] = -1e30f; l[r] = 0.f; }
  const float scale = 0.125f;

  // prime buffer 0
  dma_tile(Kbh + (size_t)0 * BN * DKDIM, 128, &sK[0][0], t);
  dma_tile(Vbh + 0 * BN, (size_t)S_LEN * 2, &sV[0][0], t);

  for (int kt = 0; kt < NT; ++kt) {
    const int cur = kt & 1;
    __syncthreads();  // all waves done computing on the other buffer
    if (kt + 1 < NT) {
      dma_tile(Kbh + (size_t)(kt + 1) * BN * DKDIM, 128, &sK[cur ^ 1][0], t);
      dma_tile(Vbh + (kt + 1) * BN, (size_t)S_LEN * 2, &sV[cur ^ 1][0], t);
      WAIT_ASYNC(8);   // async completes in order -> tile kt's 8 ops are done
    } else {
      WAIT_ASYNC(0);
    }
    __syncthreads();  // tile kt visible to all waves

    const _Float16* sKb = &sK[cur][0];
    const _Float16* sVb = &sV[cur][0];

    // ---- S = Q K^T ----
    v8f sf[4];
#pragma unroll
    for (int nt = 0; nt < 4; ++nt) {
      const _Float16* krow = sKb + (nt * 16 + llo) * LDK;
      v8h x0 = *(const v8h*)(krow + 8 * lhi);
      v8h x1 = *(const v8h*)(krow + 8 * lhi + 16);
      v8h y0 = *(const v8h*)(krow + 32 + 8 * lhi);
      v8h y1 = *(const v8h*)(krow + 32 + 8 * lhi + 16);
      v16h bk0, bk1;
#pragma unroll
      for (int e = 0; e < 8; ++e) {
        bk0[e] = x0[e]; bk0[8 + e] = x1[e];
        bk1[e] = y0[e]; bk1[8 + e] = y1[e];
      }
      v8f acc = (v8f)(0.f);
      acc = __builtin_amdgcn_wmma_f32_16x16x32_f16(false, aq[0], false, bk0, (short)0, acc, false, false);
      acc = __builtin_amdgcn_wmma_f32_16x16x32_f16(false, aq[1], false, bk1, (short)0, acc, false, false);
      sf[nt] = acc;
    }

    // ---- online softmax ----
    float alpha[8];
#pragma unroll
    for (int r = 0; r < 8; ++r) {
      float v = fmaxf(fmaxf(sf[0][r], sf[1][r]), fmaxf(sf[2][r], sf[3][r])) * scale;
      v = red16_max(v);
      float mn = fmaxf(m[r], v);
      alpha[r] = __expf(m[r] - mn);
      m[r] = mn;
      l[r] *= alpha[r];
    }
#pragma unroll
    for (int nt = 0; nt < 4; ++nt)
#pragma unroll
      for (int r = 0; r < 8; ++r) oacc[nt][r] *= alpha[r];

    // ---- p = exp(scale*s - m) * w_key ; C/D -> A re-layout via per-wave LDS tile ----
    float rs[8];
#pragma unroll
    for (int r = 0; r < 8; ++r) rs[r] = 0.f;
    _Float16* pw = &sP[wave][0];
#pragma unroll
    for (int nt = 0; nt < 4; ++nt) {
      const float w = sW[kt * BN + nt * 16 + llo];
#pragma unroll
      for (int r = 0; r < 8; ++r) {
        float p = __expf(sf[nt][r] * scale - m[r]) * w;
        rs[r] += p;
        pw[(r + 8 * lhi) * LDK + nt * 16 + llo] = (h16)p;
      }
    }
#pragma unroll
    for (int r = 0; r < 8; ++r) l[r] += red16_sum(rs[r]);

    asm volatile("s_wait_dscnt 0" ::: "memory");

    // ---- P A-fragments ----
    v16h pa0, pa1;
    {
      const _Float16* pr = &sP[wave][llo * LDK];
      v8h a0 = *(const v8h*)(pr + 8 * lhi);
      v8h a1 = *(const v8h*)(pr + 8 * lhi + 16);
      v8h b0 = *(const v8h*)(pr + 32 + 8 * lhi);
      v8h b1 = *(const v8h*)(pr + 32 + 8 * lhi + 16);
#pragma unroll
      for (int e = 0; e < 8; ++e) {
        pa0[e] = a0[e]; pa0[8 + e] = a1[e];
        pa1[e] = b0[e]; pa1[8 + e] = b1[e];
      }
    }

    // ---- O += P V ----
#pragma unroll
    for (int nt = 0; nt < 4; ++nt) {
      const _Float16* vr = sVb + (nt * 16 + llo) * LDK;
      v8h x0 = *(const v8h*)(vr + 8 * lhi);
      v8h x1 = *(const v8h*)(vr + 8 * lhi + 16);
      v8h y0 = *(const v8h*)(vr + 32 + 8 * lhi);
      v8h y1 = *(const v8h*)(vr + 32 + 8 * lhi + 16);
      v16h bv0, bv1;
#pragma unroll
      for (int e = 0; e < 8; ++e) {
        bv0[e] = x0[e]; bv0[8 + e] = x1[e];
        bv1[e] = y0[e]; bv1[8 + e] = y1[e];
      }
      oacc[nt] = __builtin_amdgcn_wmma_f32_16x16x32_f16(false, pa0, false, bv0, (short)0, oacc[nt], false, false);
      oacc[nt] = __builtin_amdgcn_wmma_f32_16x16x32_f16(false, pa1, false, bv1, (short)0, oacc[nt], false, false);
    }
  }

  float inv[8];
#pragma unroll
  for (int r = 0; r < 8; ++r) inv[r] = 1.0f / l[r];
#pragma unroll
  for (int nt = 0; nt < 4; ++nt)
#pragma unroll
    for (int r = 0; r < 8; ++r) {
      const int row = qt * BM + wave * 16 + 8 * lhi + r;
      Ob[(size_t)row * DKDIM + 16 * nt + llo] = oacc[nt][r] * inv[r];
    }
}

// ============ fallback (self-contained, f32 inputs; used when ws is too small) ============
__global__ __launch_bounds__(128) void trapz_flash_attn(
    const float* __restrict__ Q, const float* __restrict__ K,
    const float* __restrict__ V, const float* __restrict__ C,
    float* __restrict__ O)
{
  __shared__ __align__(32) _Float16 sK[BN * LDK];
  __shared__ __align__(32) _Float16 sV[DKDIM * LDK];
  __shared__ float sW[S_LEN];
  __shared__ __align__(32) unsigned char sX[SCRATCH_BYTES];
  _Float16* sP = (_Float16*)sX;
#if USE_ASYNC_LDS
  float* sKf = (float*)sX;
  float* sVf = (float*)(sX + 16384);
#endif

  const int t    = threadIdx.x;
  const int wave = t >> 5;
  const int lane = t & 31;
  const int lhi  = lane >> 4;
  const int llo  = lane & 15;
  const int qt = blockIdx.x & (S_LEN / BM - 1);
  const int bh = blockIdx.x / (S_LEN / BM);

  const float* Qb = Q + (size_t)bh * S_LEN * DKDIM;
  const float* Kb = K + (size_t)bh * S_LEN * DKDIM;
  const float* Vb = V + (size_t)bh * S_LEN * DKDIM;
  float*       Ob = O + (size_t)bh * S_LEN * DKDIM;

  for (int s = t; s < S_LEN; s += 128) {
    float cm = (s > 0)         ? fabsf(C[s] - C[s - 1]) : 0.f;
    float cp = (s < S_LEN - 1) ? fabsf(C[s + 1] - C[s]) : 0.f;
    sW[s] = 0.5f * (cm + cp);
  }

  const int qrow = qt * BM + wave * 16 + llo;
  v16h aq[2];
#pragma unroll
  for (int c = 0; c < 2; ++c) {
    const float* qp = Qb + qrow * DKDIM + 32 * c + 8 * lhi;
    v4f f0 = *(const v4f*)(qp);
    v4f f1 = *(const v4f*)(qp + 4);
    v4f f2 = *(const v4f*)(qp + 16);
    v4f f3 = *(const v4f*)(qp + 20);
#pragma unroll
    for (int i = 0; i < 4; ++i) {
      aq[c][i]      = (h16)f0[i];
      aq[c][4 + i]  = (h16)f1[i];
      aq[c][8 + i]  = (h16)f2[i];
      aq[c][12 + i] = (h16)f3[i];
    }
  }

  v8f oacc[4];
  float m[8], l[8];
#pragma unroll
  for (int i = 0; i < 4; ++i) oacc[i] = (v8f)(0.f);
#pragma unroll
  for (int r = 0; r < 8; ++r) { m[r] = -1e30f; l[r] = 0.f; }
  const float scale = 0.125f;

  for (int kt = 0; kt < NT; ++kt) {
    __syncthreads();
#if USE_ASYNC_LDS
    {
      const char* gk = (const char*)(Kb + (size_t)kt * BN * DKDIM);
      const char* gv = (const char*)(Vb + (size_t)kt * BN * DKDIM);
#pragma unroll
      for (int i = 0; i < 8; ++i) {
        const unsigned off = (unsigned)(i * 128 + t) * 16;
        __builtin_amdgcn_global_load_async_to_lds_b128(
            (gb128_p)(uintptr_t)(gk + off),
            (lb128_p)(unsigned)(uintptr_t)((char*)sKf + off), 0, 0);
        __builtin_amdgcn_global_load_async_to_lds_b128(
            (gb128_p)(uintptr_t)(gv + off),
            (lb128_p)(unsigned)(uintptr_t)((char*)sVf + off), 0, 0);
      }
      WAIT_ASYNC(0);
    }
    __syncthreads();
#endif
    {
      const int row = t >> 1, d0 = (t & 1) * 32;
#if USE_ASYNC_LDS
      const v4f* kg = (const v4f*)(sKf + row * DKDIM + d0);
      const v4f* vg = (const v4f*)(sVf + row * DKDIM + d0);
#else
      const v4f* kg = (const v4f*)(Kb + (size_t)(kt * BN + row) * DKDIM + d0);
      const v4f* vg = (const v4f*)(Vb + (size_t)(kt * BN + row) * DKDIM + d0);
#endif
      _Float16* dst = &sK[row * LDK + d0];
#pragma unroll
      for (int i = 0; i < 4; ++i) {
        v4f a = kg[2 * i], b = kg[2 * i + 1];
        v8h h;
#pragma unroll
        for (int e = 0; e < 4; ++e) { h[e] = (h16)a[e]; h[4 + e] = (h16)b[e]; }
        *(v8h*)(dst + 8 * i) = h;
      }
#pragma unroll
      for (int i = 0; i < 8; ++i) {
        v4f a = vg[i];
#pragma unroll
        for (int e = 0; e < 4; ++e)
          sV[(d0 + 4 * i + e) * LDK + row] = (h16)a[e];
      }
    }
    __syncthreads();

    v8f sf[4];
#pragma unroll
    for (int nt = 0; nt < 4; ++nt) {
      const _Float16* krow = &sK[(nt * 16 + llo) * LDK];
      v8h x0 = *(const v8h*)(krow + 8 * lhi);
      v8h x1 = *(const v8h*)(krow + 8 * lhi + 16);
      v8h y0 = *(const v8h*)(krow + 32 + 8 * lhi);
      v8h y1 = *(const v8h*)(krow + 32 + 8 * lhi + 16);
      v16h bk0, bk1;
#pragma unroll
      for (int e = 0; e < 8; ++e) {
        bk0[e] = x0[e]; bk0[8 + e] = x1[e];
        bk1[e] = y0[e]; bk1[8 + e] = y1[e];
      }
      v8f acc = (v8f)(0.f);
      acc = __builtin_amdgcn_wmma_f32_16x16x32_f16(false, aq[0], false, bk0, (short)0, acc, false, false);
      acc = __builtin_amdgcn_wmma_f32_16x16x32_f16(false, aq[1], false, bk1, (short)0, acc, false, false);
      sf[nt] = acc;
    }

    float alpha[8];
#pragma unroll
    for (int r = 0; r < 8; ++r) {
      float v = fmaxf(fmaxf(sf[0][r], sf[1][r]), fmaxf(sf[2][r], sf[3][r])) * scale;
      v = red16_max(v);
      float mn = fmaxf(m[r], v);
      alpha[r] = __expf(m[r] - mn);
      m[r] = mn;
      l[r] *= alpha[r];
    }
#pragma unroll
    for (int nt = 0; nt < 4; ++nt)
#pragma unroll
      for (int r = 0; r < 8; ++r) oacc[nt][r] *= alpha[r];

    float rs[8];
#pragma unroll
    for (int r = 0; r < 8; ++r) rs[r] = 0.f;
    _Float16* pw = sP + wave * 16 * LDK;
#pragma unroll
    for (int nt = 0; nt < 4; ++nt) {
      const float w = sW[kt * BN + nt * 16 + llo];
#pragma unroll
      for (int r = 0; r < 8; ++r) {
        float p = __expf(sf[nt][r] * scale - m[r]) * w;
        rs[r] += p;
        pw[(r + 8 * lhi) * LDK + nt * 16 + llo] = (h16)p;
      }
    }
#pragma unroll
    for (int r = 0; r < 8; ++r) l[r] += red16_sum(rs[r]);

    asm volatile("s_wait_dscnt 0" ::: "memory");

    v16h pa0, pa1;
    {
      const _Float16* pr = sP + wave * 16 * LDK + llo * LDK;
      v8h a0 = *(const v8h*)(pr + 8 * lhi);
      v8h a1 = *(const v8h*)(pr + 8 * lhi + 16);
      v8h b0 = *(const v8h*)(pr + 32 + 8 * lhi);
      v8h b1 = *(const v8h*)(pr + 32 + 8 * lhi + 16);
#pragma unroll
      for (int e = 0; e < 8; ++e) {
        pa0[e] = a0[e]; pa0[8 + e] = a1[e];
        pa1[e] = b0[e]; pa1[8 + e] = b1[e];
      }
    }
#pragma unroll
    for (int nt = 0; nt < 4; ++nt) {
      const _Float16* vr = &sV[(nt * 16 + llo) * LDK];
      v8h x0 = *(const v8h*)(vr + 8 * lhi);
      v8h x1 = *(const v8h*)(vr + 8 * lhi + 16);
      v8h y0 = *(const v8h*)(vr + 32 + 8 * lhi);
      v8h y1 = *(const v8h*)(vr + 32 + 8 * lhi + 16);
      v16h bv0, bv1;
#pragma unroll
      for (int e = 0; e < 8; ++e) {
        bv0[e] = x0[e]; bv0[8 + e] = x1[e];
        bv1[e] = y0[e]; bv1[8 + e] = y1[e];
      }
      oacc[nt] = __builtin_amdgcn_wmma_f32_16x16x32_f16(false, pa0, false, bv0, (short)0, oacc[nt], false, false);
      oacc[nt] = __builtin_amdgcn_wmma_f32_16x16x32_f16(false, pa1, false, bv1, (short)0, oacc[nt], false, false);
    }
  }

  float inv[8];
#pragma unroll
  for (int r = 0; r < 8; ++r) inv[r] = 1.0f / l[r];
#pragma unroll
  for (int nt = 0; nt < 4; ++nt)
#pragma unroll
    for (int r = 0; r < 8; ++r) {
      const int row = qt * BM + wave * 16 + 8 * lhi + r;
      Ob[(size_t)row * DKDIM + 16 * nt + llo] = oacc[nt][r] * inv[r];
    }
}

extern "C" void kernel_launch(void* const* d_in, const int* in_sizes, int n_in,
                              void* d_out, int out_size, void* d_ws, size_t ws_size,
                              hipStream_t stream) {
  const float* q = (const float*)d_in[0];
  const float* k = (const float*)d_in[1];
  const float* v = (const float*)d_in[2];
  const float* c = (const float*)d_in[3];
  float* o = (float*)d_out;

  const size_t khBytes = (size_t)NBH * S_LEN * DKDIM * sizeof(h16);  // 4 MB
  dim3 grid(NBH * (S_LEN / BM));   // 512 WGs
  dim3 block(128);                 // 4 wave32

  if (ws_size >= 2 * khBytes) {
    h16* Kh  = (h16*)d_ws;
    h16* Vth = (h16*)((char*)d_ws + khBytes);
    convert_kv<<<dim3(NBH * NT), block, 0, stream>>>(k, v, Kh, Vth);
    trapz_flash_attn_pipe<<<grid, block, 0, stream>>>(q, Kh, Vth, c, o);
  } else {
    trapz_flash_attn<<<grid, block, 0, stream>>>(q, k, v, c, o);
  }
}
